// ChannelWiseFC_7456063226005
// MI455X (gfx1250) — matrix operand order repaired and checked by reading, not compiled, not run on hardware
//
#include <hip/hip_runtime.h>
#include <stdint.h>

// ---------- CDNA5 WMMA / TDM types ----------
typedef __attribute__((ext_vector_type(16))) _Float16 v16h;
typedef __attribute__((ext_vector_type(8)))  float    v8f;
typedef __attribute__((ext_vector_type(4)))  _Float16 h4;
typedef __attribute__((ext_vector_type(4)))  float    f32x4;
typedef __attribute__((ext_vector_type(4)))  unsigned int u32x4;
typedef __attribute__((ext_vector_type(8)))  int      i32x8;
typedef __attribute__((ext_vector_type(4)))  int      i32x4;

// ---------- problem constants (match reference) ----------
constexpr int kB = 8, kC = 32, kN = 1024, kF = 512;
constexpr int TILE_R = 64;    // output rows per workgroup
constexpr int TILE_F = 128;   // output cols per workgroup
constexpr int TILE_K = 32;    // K per WMMA step (f16 WMMA K)
constexpr int KSTEPS = kF / TILE_K;  // 16
constexpr int AS     = 40;    // converted-LDS row stride in halves (80B, conflict-free)

union V16 { v16h v; f32x4 f4[2]; };

// low 32 bits of a generic pointer to LDS == LDS byte offset (ISA flat-addressing rule)
__device__ inline uint32_t lds_addr32(const void* p) {
  return (uint32_t)(uintptr_t)p;
}

// Issue one TDM 2D tile load: `rows` rows of TILE_K f32, source row stride kF f32.
// D# per CDNA5 ISA 8.3/8.4: group0 = {flags, lds_addr, global_addr, type=2},
// group1 = {data_size, tensor dims, tile dims, dim0 stride}; groups 2/3 zero (2D tile).
__device__ inline void tdm_load_tile_2d(uint32_t lds_off, const float* gsrc, int rows) {
  uint64_t ga = (uint64_t)(uintptr_t)gsrc;
  u32x4 g0;
  g0[0] = 1u;                                           // count=1, no gather, user desc
  g0[1] = lds_off;                                      // LDS byte address
  g0[2] = (uint32_t)ga;                                 // global_addr[31:0]
  g0[3] = (uint32_t)((ga >> 32) & 0x1FFFFFFull)         // global_addr[56:32]
        | (2u << 30);                                   // type = 2 ("image")
  i32x8 g1;
  g1[0] = (int)(2u << 16);                              // workgroup_mask=0, data_size=4B
  g1[1] = (int)(((unsigned)TILE_K & 0xFFFFu) << 16);    // tensor_dim0[15:0]=32
  g1[2] = (int)(((unsigned)rows & 0xFFFFu) << 16);      // tensor_dim0[31:16]=0 | tensor_dim1[15:0]
  g1[3] = (int)((unsigned)TILE_K << 16);                // tensor_dim1[31:16]=0 | tile_dim0=32
  g1[4] = rows;                                         // tile_dim1 | tile_dim2=0 (2D)
  g1[5] = kF;                                           // tensor_dim0_stride[31:0] = 512
  g1[6] = 0;                                            // stride0[47:32]=0 | stride1[15:0]=0
  g1[7] = 0;                                            // stride1[47:16]=0
  i32x4 z4 = {0, 0, 0, 0};
  i32x8 z8 = {0, 0, 0, 0, 0, 0, 0, 0};
  // 6-arg toolchain variant: (g0, g1, g2, g3, extra, cpol)
  __builtin_amdgcn_tensor_load_to_lds(g0, g1, z4, z4, z8, 0 /*cpol*/);
}

// split one f32x4 into f16 hi/lo residual pairs and store 8B each to LDS
__device__ inline void cvt_store4(_Float16* ph, _Float16* pl, f32x4 s) {
  _Float16 h0 = (_Float16)s.x, h1 = (_Float16)s.y,
           h2 = (_Float16)s.z, h3 = (_Float16)s.w;
  h4 hv = {h0, h1, h2, h3};
  h4 lv = {(_Float16)(s.x - (float)h0), (_Float16)(s.y - (float)h1),
           (_Float16)(s.z - (float)h2), (_Float16)(s.w - (float)h3)};
  *(h4*)ph = hv;
  *(h4*)pl = lv;
}

__global__ __launch_bounds__(256) void ChannelWiseFC_wmma(
    const float* __restrict__ x,     // [B, C, N, F]
    const float* __restrict__ w,     // [C, F, F]  (out = x @ w^T + bias)
    const float* __restrict__ bias,  // [C, F]
    float* __restrict__ out)         // [B, C, N, F]
{
  // raw f32 tiles filled by the Tensor Data Mover (dense row-major)
  __shared__ float   Araw[2][TILE_R * TILE_K];   // 2 x 8 KB
  __shared__ float   Braw[2][TILE_F * TILE_K];   // 2 x 16 KB
  // converted f16 hi/lo tiles (padded rows, WMMA operand friendly)
  __shared__ _Float16 Ah[2][TILE_R * AS];
  __shared__ _Float16 Al[2][TILE_R * AS];
  __shared__ _Float16 Bh[2][TILE_F * AS];
  __shared__ _Float16 Bl[2][TILE_F * AS];

  const int c  = blockIdx.z;
  const int r0 = blockIdx.y * TILE_R;        // global row within channel (B*N flattened)
  const int f0 = blockIdx.x * TILE_F;
  const int bi = r0 >> 10;                   // batch index (TILE_R divides N)
  const int n0 = r0 & (kN - 1);

  const float* xb = x + ((size_t)(bi * kC + c) * kN + n0) * kF;   // row stride kF
  const float* wb = w + (size_t)c * kF * kF + (size_t)f0 * kF;    // row stride kF

  const int t = threadIdx.x;

  // ---- raw-LDS -> converted-LDS pass (each element converted exactly once) ----
  // A tile: 512 f32x4 slots, 2/thread; B tile: 1024 slots, 4/thread.
  auto convert_tiles = [&](int buf) {
#pragma unroll
    for (int i = 0; i < 2; ++i) {
      int s = t + i * 256, row = s >> 3, q = s & 7;
      f32x4 v = *(const f32x4*)&Araw[buf][s * 4];
      cvt_store4(&Ah[buf][row * AS + q * 4], &Al[buf][row * AS + q * 4], v);
    }
#pragma unroll
    for (int i = 0; i < 4; ++i) {
      int s = t + i * 256, row = s >> 3, q = s & 7;
      f32x4 v = *(const f32x4*)&Braw[buf][s * 4];
      cvt_store4(&Bh[buf][row * AS + q * 4], &Bl[buf][row * AS + q * 4], v);
    }
  };

  // ---- wave/lane geometry ----
  const int wv   = t >> 5;       // wave 0..7
  const int lane = t & 31;
  const int rg   = wv & 3;       // row group: rows [rg*16, rg*16+16)
  const int cg   = wv >> 2;      // col group: cols [cg*64, cg*64+64)
  const int h    = lane >> 4;    // lane half (CDNA5 operand layout)
  const int lm   = lane & 15;

  v8f zero = {0.f, 0.f, 0.f, 0.f, 0.f, 0.f, 0.f, 0.f};
  v8f acc[4] = {zero, zero, zero, zero};

  // ---- prologue: TDM fetch of tile 0 (wave 0 drives the DMA engine) ----
  if (wv == 0) {
    tdm_load_tile_2d(lds_addr32(&Araw[0][0]), xb, TILE_R);
    tdm_load_tile_2d(lds_addr32(&Braw[0][0]), wb, TILE_F);
  }

  int buf = 0;
  for (int ks = 0; ks < KSTEPS; ++ks) {
    if (wv == 0) __builtin_amdgcn_s_wait_tensorcnt(0);  // raw[buf] transfer complete
    __syncthreads();                                    // all waves: raw[buf] ready

    // kick off DMA for the next K-step; overlaps convert + WMMA below.
    // raw[buf^1]'s last readers finished before the barrier above.
    if (ks + 1 < KSTEPS && wv == 0) {
      const int kb = (ks + 1) * TILE_K;
      tdm_load_tile_2d(lds_addr32(&Araw[buf ^ 1][0]), xb + kb, TILE_R);
      tdm_load_tile_2d(lds_addr32(&Braw[buf ^ 1][0]), wb + kb, TILE_F);
    }

    convert_tiles(buf);
    __syncthreads();                                    // converted tiles visible

    // A operand: row = rg*16+lm; elements [h*8, h*8+8) and [16+h*8, 16+h*8+8)
    const int arow = rg * 16 + lm;
    V16 aH, aL;
    aH.f4[0] = *(const f32x4*)&Ah[buf][arow * AS + h * 8];
    aH.f4[1] = *(const f32x4*)&Ah[buf][arow * AS + 16 + h * 8];
    aL.f4[0] = *(const f32x4*)&Al[buf][arow * AS + h * 8];
    aL.f4[1] = *(const f32x4*)&Al[buf][arow * AS + 16 + h * 8];

#pragma unroll
    for (int nt = 0; nt < 4; ++nt) {
      const int brow = cg * 64 + nt * 16 + lm;   // B elements [h*8, h*8+8) and [16+h*8, 16+h*8+8) at col n=lm
      V16 bH, bL;
      bH.f4[0] = *(const f32x4*)&Bh[buf][brow * AS + h * 8];
      bH.f4[1] = *(const f32x4*)&Bh[buf][brow * AS + 16 + h * 8];
      bL.f4[0] = *(const f32x4*)&Bl[buf][brow * AS + h * 8];
      bL.f4[1] = *(const f32x4*)&Bl[buf][brow * AS + 16 + h * 8];

      // fp32-emulating split: hi*hi + hi*lo + lo*hi, f32 accumulate
      acc[nt] = __builtin_amdgcn_wmma_f32_16x16x32_f16(
          false, aH.v, false, bH.v, (short)0, acc[nt], false, false);
      acc[nt] = __builtin_amdgcn_wmma_f32_16x16x32_f16(
          false, aH.v, false, bL.v, (short)0, acc[nt], false, false);
      acc[nt] = __builtin_amdgcn_wmma_f32_16x16x32_f16(
          false, aL.v, false, bH.v, (short)0, acc[nt], false, false);
    }

    buf ^= 1;
  }

  // ---- epilogue: out = acc + bias; C/D layout: lane -> N=lm, VGPR v -> M = v + 8h ----
  float* ob = out + ((size_t)(bi * kC + c) * kN + n0) * kF;
  const float* bc = bias + (size_t)c * kF;
#pragma unroll
  for (int nt = 0; nt < 4; ++nt) {
    const int f  = f0 + cg * 64 + nt * 16 + lm;
    const float bv = bc[f];
#pragma unroll
    for (int v = 0; v < 8; ++v) {
      const int row = rg * 16 + h * 8 + v;
      ob[(size_t)row * kF + f] = acc[nt][v] + bv;
    }
  }
}

extern "C" void kernel_launch(void* const* d_in, const int* in_sizes, int n_in,
                              void* d_out, int out_size, void* d_ws, size_t ws_size,
                              hipStream_t stream) {
  const float* x    = (const float*)d_in[0];
  const float* w    = (const float*)d_in[1];
  const float* bias = (const float*)d_in[2];
  float* out        = (float*)d_out;

  dim3 grid(kF / TILE_F, (kB * kN) / TILE_R, kC);  // (4, 128, 32)
  dim3 block(256);
  hipLaunchKernelGGL(ChannelWiseFC_wmma, grid, block, 0, stream, x, w, bias, out);
}
